// HybridCNNGRUSNN_55954833933055
// MI455X (gfx1250) — matrix operand order repaired and checked by reading, not compiled, not run on hardware
//
#include <hip/hip_runtime.h>

// ---------------------------------------------------------------------------
// Types for CDNA5 WMMA (wave32)
// ---------------------------------------------------------------------------
typedef __attribute__((ext_vector_type(8)))  float  v8f;
typedef __attribute__((ext_vector_type(8)))  __bf16 v8bf;
typedef __attribute__((ext_vector_type(16))) __bf16 v16bf;
typedef int v4i_async __attribute__((vector_size(16)));   // matches async builtin proto

// gfx1250 async global->LDS path (ASYNCcnt), with portable fallback
#if defined(__has_builtin)
#  if __has_builtin(__builtin_amdgcn_global_load_async_to_lds_b128) && \
      __has_builtin(__builtin_amdgcn_s_wait_asynccnt)
#    define ATHENA_ASYNC_LDS 1
#  endif
#endif
#ifndef ATHENA_ASYNC_LDS
#  define ATHENA_ASYNC_LDS 0
#endif

// ---------------------------------------------------------------------------
// Generic bf16 WMMA GEMM:  C[M,N] = act(alpha[m]*(A·Bt^T) + beta[m])
//   A : [Mp, K]  bf16 row-major (K contiguous), Mp mult of 16, K mult of 64
//   Bt: [Np, K]  bf16 row-major (B transposed), Np mult of 64
// Block = 4 waves; block tile 16(M) x 64(N). The shared 16xK A tile is staged
// through a double-buffered LDS tile with async global->LDS copies; B operand
// streams straight from global (b128 clauses + prefetch).
// mode 0: C[n*ldc + m] (+ optional bf16 mirror)   mode 1: NCHW conv store
// ---------------------------------------------------------------------------
__global__ void k_gemm_bf16(const __bf16* __restrict__ A,
                            const __bf16* __restrict__ Bt,
                            float* __restrict__ C,
                            __bf16* __restrict__ Cbf,
                            int K, int ldc, int Mstore, int Nstore,
                            const float* __restrict__ alpha,
                            const float* __restrict__ beta,
                            int act, int mode, int HW)
{
    __shared__ __bf16 lA[2][16 * 64];   // double-buffered 16x64 A stage (4 KB)

    const int lane = threadIdx.x;        // 0..31
    const int h    = lane >> 4;          // lane half
    const int lr   = lane & 15;
    const int tid  = threadIdx.y * 32 + lane;   // 0..127
    const size_t m0 = (size_t)blockIdx.y * 16u;
    const size_t n0 = ((size_t)blockIdx.x * 4u + threadIdx.y) * 16u;

    const __bf16* Bp = Bt + (n0 + lr) * (size_t)K;

    // one 16B chunk per thread per stage: 128 threads x 16B = 16x64 bf16 tile
    const int srow = tid >> 3;           // 0..15
    const int scol = (tid & 7) << 3;     // 0,8,...,56 (halfs)
    auto stage = [&](int buf, int kc) {
        const __bf16* g = A + (m0 + srow) * (size_t)K + kc + scol;
        __bf16* l = &lA[buf][(srow << 6) + scol];
#if ATHENA_ASYNC_LDS
        __builtin_amdgcn_global_load_async_to_lds_b128(
            (__attribute__((address_space(1))) v4i_async*)g,
            (__attribute__((address_space(3))) v4i_async*)l, 0, 0);
#else
        *(v8bf*)l = *(const v8bf*)g;
#endif
    };

    v8f acc = {};
    stage(0, 0);
    int pb = 0;
    for (int kc = 0; kc < K; kc += 64) {
        const bool more = (kc + 64 < K);
        if (more) stage(pb ^ 1, kc + 64);
#if ATHENA_ASYNC_LDS
        if (more) __builtin_amdgcn_s_wait_asynccnt(1);   // stage kc complete
        else      __builtin_amdgcn_s_wait_asynccnt(0);
#endif
        __syncthreads();                                  // stage visible to all waves
        const __bf16* Ab = &lA[pb][lr << 6];
        #pragma unroll
        for (int s = 0; s < 2; ++s) {
            const int kk = s * 32;
            // A 16x32 bf16 frag: lane half h holds K offsets {8h.., 16+8h..} (LDS)
            v8bf a0 = *(const v8bf*)(Ab + kk + 8 * h);
            v8bf a1 = *(const v8bf*)(Ab + kk + 16 + 8 * h);
            // B 32x16 bf16 frag: lane half h holds K offsets {8h.., 16+8h..}
            v8bf b0 = *(const v8bf*)(Bp + kc + kk + 8 * h);
            v8bf b1 = *(const v8bf*)(Bp + kc + kk + 16 + 8 * h);
            __builtin_prefetch(Bp + kc + kk + 128, 0, 0);
            v16bf av = __builtin_shufflevector(a0, a1, 0,1,2,3,4,5,6,7,8,9,10,11,12,13,14,15);
            v16bf bv = __builtin_shufflevector(b0, b1, 0,1,2,3,4,5,6,7,8,9,10,11,12,13,14,15);
            acc = __builtin_amdgcn_wmma_f32_16x16x32_bf16(false, av, false, bv,
                                                          (short)0, acc, false, false);
        }
        __syncthreads();                                  // readers done before reuse
        pb ^= 1;
    }

    const int n = (int)n0 + lr;
    if (n >= Nstore) return;
    #pragma unroll
    for (int r = 0; r < 8; ++r) {
        int m = (int)m0 + r + 8 * h;     // C/D layout: VGPR r -> M = r + 8*half
        if (m >= Mstore) continue;
        float v = acc[r];
        if (alpha) v *= alpha[m];
        if (beta)  v += beta[m];
        if (act == 1)      v = v > 0.f ? v : 0.f;
        else if (act == 2) v = v > 0.f ? v : 0.1f * v;
        if (mode == 0) {
            C[(size_t)n * ldc + m] = v;
            if (Cbf) Cbf[(size_t)n * ldc + m] = (__bf16)v;
        } else {
            int b = n / HW, yx = n - b * HW;
            C[((size_t)b * Mstore + m) * HW + yx] = v;
        }
    }
}

// ---------------------------------------------------------------------------
// fp32 -> bf16 pack with optional M/K zero padding (src [M,K] -> dst [Mp,Kp])
// ---------------------------------------------------------------------------
__global__ void k_pack_bf16(const float* __restrict__ src, __bf16* __restrict__ dst,
                            int M, int K, int Mp, int Kp)
{
    size_t i = (size_t)blockIdx.x * blockDim.x + threadIdx.x;
    size_t total = (size_t)Mp * Kp;
    if (i >= total) return;
    int k = (int)(i % Kp);
    int m = (int)(i / Kp);
    float v = (m < M && k < K) ? src[(size_t)m * K + k] : 0.f;
    dst[i] = (__bf16)v;
}

// ---------------------------------------------------------------------------
// im2col (3x3, pad=1) NCHW fp32 -> bf16 [B*H*W, Kp] rows, K = C*9 zero-padded
// ---------------------------------------------------------------------------
__global__ void k_im2col(const float* __restrict__ src, __bf16* __restrict__ dst,
                         int Bn, int C, int H, int W, int Kp)
{
    size_t i = (size_t)blockIdx.x * blockDim.x + threadIdx.x;
    size_t total = (size_t)Bn * H * W * Kp;
    if (i >= total) return;
    int k = (int)(i % Kp);
    size_t n = i / Kp;
    int x = (int)(n % W);
    int y = (int)((n / W) % H);
    int b = (int)(n / ((size_t)W * H));
    float v = 0.f;
    if (k < C * 9) {
        int ci = k / 9, r = k % 9;
        int yy = y + r / 3 - 1, xx = x + r % 3 - 1;
        if (yy >= 0 && yy < H && xx >= 0 && xx < W)
            v = src[(((size_t)b * C + ci) * H + yy) * W + xx];
    }
    dst[i] = (__bf16)v;
}

// ---------------------------------------------------------------------------
// 2x2 stride-2 max pool, NCHW fp32
// ---------------------------------------------------------------------------
__global__ void k_maxpool2(const float* __restrict__ src, float* __restrict__ dst,
                           int BC, int H, int W)
{
    int Ho = H >> 1, Wo = W >> 1;
    size_t i = (size_t)blockIdx.x * blockDim.x + threadIdx.x;
    size_t total = (size_t)BC * Ho * Wo;
    if (i >= total) return;
    int x = (int)(i % Wo);
    int y = (int)((i / Wo) % Ho);
    size_t bc = i / ((size_t)Wo * Ho);
    const float* s = src + (bc * H + 2 * y) * (size_t)W + 2 * x;
    dst[i] = fmaxf(fmaxf(s[0], s[1]), fmaxf(s[W], s[W + 1]));
}

// ---------------------------------------------------------------------------
// Precompute BN affine: alpha = g*rsqrt(v+eps), beta = alpha*(convb - rm) + bb
// ---------------------------------------------------------------------------
__global__ void k_bnparam(const float* g, const float* bb, const float* rm,
                          const float* rv, const float* convb,
                          float* alpha, float* beta, int C)
{
    int c = blockIdx.x * blockDim.x + threadIdx.x;
    if (c >= C) return;
    float s = g[c] * rsqrtf(rv[c] + 1e-5f);
    alpha[c] = s;
    beta[c]  = s * (convb[c] - rm[c]) + bb[c];
}

// ---------------------------------------------------------------------------
// Broadcast p2 [B,F] over time -> bf16 [B*T, F], row n = b*T + t
// ---------------------------------------------------------------------------
__global__ void k_broadcast_time(const float* __restrict__ p2, __bf16* __restrict__ xt,
                                 int Bn, int Tn, int F)
{
    size_t i = (size_t)blockIdx.x * blockDim.x + threadIdx.x;
    size_t total = (size_t)Bn * Tn * F;
    if (i >= total) return;
    int j = (int)(i % F);
    int b = (int)(i / ((size_t)F * Tn));
    xt[i] = (__bf16)p2[(size_t)b * F + j];
}

__global__ void k_fill_f32(float* p, size_t n, float v)
{
    size_t i = (size_t)blockIdx.x * blockDim.x + threadIdx.x;
    if (i < n) p[i] = v;
}
__global__ void k_fill_bf16(__bf16* p, size_t n, float v)
{
    size_t i = (size_t)blockIdx.x * blockDim.x + threadIdx.x;
    if (i < n) p[i] = (__bf16)v;
}

// ---------------------------------------------------------------------------
// GRU gate step (PyTorch gate order r,z,n). gi/gh already include biases.
// Writes h (fp32 + bf16 for next GEMM) and y[b][tt][dir*H + j].
// ---------------------------------------------------------------------------
__global__ void k_gru_step(const float* __restrict__ gi, const float* __restrict__ gh,
                           float* __restrict__ hf, __bf16* __restrict__ hbf,
                           float* __restrict__ y, int tt, int Bn, int H, int dir, int Tn)
{
    int i = blockIdx.x * blockDim.x + threadIdx.x;
    if (i >= Bn * H) return;
    int b = i / H, j = i - b * H;
    const float* g  = gi + ((size_t)b * Tn + tt) * (3 * H);
    const float* gg = gh + (size_t)b * (3 * H);
    float r  = 1.f / (1.f + __expf(-(g[j]         + gg[j])));
    float z  = 1.f / (1.f + __expf(-(g[H + j]     + gg[H + j])));
    float nn = tanhf(g[2 * H + j] + r * gg[2 * H + j]);
    float hv = hf[i];
    float ho = (1.f - z) * nn + z * hv;
    hf[i]  = ho;
    hbf[i] = (__bf16)ho;
    y[((size_t)b * Tn + tt) * (2 * H) + dir * H + j] = ho;
}

// ---------------------------------------------------------------------------
// LIF scan: mem = beta*mem + inp - (mem_prev > thr); spk = (mem - thr > 0)
// ---------------------------------------------------------------------------
__global__ void k_lif(const float* __restrict__ lin, float* __restrict__ spks,
                      int Bn, int Tn, int F)
{
    int i = blockIdx.x * blockDim.x + threadIdx.x;
    if (i >= Bn * F) return;
    int b = i / F, j = i - b * F;
    float mem = 0.f;
    for (int t = 0; t < Tn; ++t) {
        size_t o = ((size_t)b * Tn + t) * F + j;
        float reset = mem > 1.0f ? 1.0f : 0.0f;
        mem = 0.95f * mem + lin[o] - reset;
        spks[o] = (mem - 1.0f) > 0.f ? 1.f : 0.f;
    }
}

// ---------------------------------------------------------------------------
// Fusion mean over T of concat(y[512], spks[256]) -> feat [B,768] (+ bf16)
// ---------------------------------------------------------------------------
__global__ void k_fusion(const float* __restrict__ y, const float* __restrict__ spks,
                         float* __restrict__ feat, __bf16* __restrict__ featbf,
                         int Bn, int Tn)
{
    int i = blockIdx.x * blockDim.x + threadIdx.x;
    if (i >= Bn * 768) return;
    int b = i / 768, j = i - b * 768;
    float s = 0.f;
    if (j < 512) {
        for (int t = 0; t < Tn; ++t) s += y[((size_t)b * Tn + t) * 512 + j];
    } else {
        for (int t = 0; t < Tn; ++t) s += spks[((size_t)b * Tn + t) * 256 + (j - 512)];
    }
    float v = s / (float)Tn;
    feat[i]   = v;
    featbf[i] = (__bf16)v;
}

// ---------------------------------------------------------------------------
// Host-side orchestration
// ---------------------------------------------------------------------------
extern "C" void kernel_launch(void* const* d_in, const int* in_sizes, int n_in,
                              void* d_out, int out_size, void* d_ws, size_t ws_size,
                              hipStream_t stream)
{
    (void)in_sizes; (void)n_in; (void)out_size; (void)ws_size;
    const int B = 64, T = 25;

    const float* x        = (const float*)d_in[0];
    const float* conv1_w  = (const float*)d_in[1];
    const float* conv1_b  = (const float*)d_in[2];
    const float* bn1_g    = (const float*)d_in[3];
    const float* bn1_b    = (const float*)d_in[4];
    const float* bn1_m    = (const float*)d_in[5];
    const float* bn1_v    = (const float*)d_in[6];
    const float* conv2_w  = (const float*)d_in[7];
    const float* conv2_b  = (const float*)d_in[8];
    const float* bn2_g    = (const float*)d_in[9];
    const float* bn2_b    = (const float*)d_in[10];
    const float* bn2_m    = (const float*)d_in[11];
    const float* bn2_v    = (const float*)d_in[12];
    const float* conv3_w  = (const float*)d_in[13];
    const float* conv3_b  = (const float*)d_in[14];
    const float* bn3_g    = (const float*)d_in[15];
    const float* bn3_b    = (const float*)d_in[16];
    const float* bn3_m    = (const float*)d_in[17];
    const float* bn3_v    = (const float*)d_in[18];
    const float* proj1_w  = (const float*)d_in[19];
    const float* proj1_b  = (const float*)d_in[20];
    const float* proj2_w  = (const float*)d_in[21];
    const float* proj2_b  = (const float*)d_in[22];
    const float* gru0_wih = (const float*)d_in[23];
    const float* gru0_whh = (const float*)d_in[24];
    const float* gru0_bih = (const float*)d_in[25];
    const float* gru0_bhh = (const float*)d_in[26];
    const float* gru1_wih = (const float*)d_in[27];
    const float* gru1_whh = (const float*)d_in[28];
    const float* gru1_bih = (const float*)d_in[29];
    const float* gru1_bhh = (const float*)d_in[30];
    const float* lifproj_w= (const float*)d_in[31];
    const float* lifproj_b= (const float*)d_in[32];
    const float* fc1_w    = (const float*)d_in[33];
    const float* fc1_b    = (const float*)d_in[34];
    const float* fc2_w    = (const float*)d_in[35];
    const float* fc2_b    = (const float*)d_in[36];
    const float* fc3_w    = (const float*)d_in[37];
    const float* fc3_b    = (const float*)d_in[38];

    char* ws = (char*)d_ws;
    const size_t MB = 1ull << 20;
    // Reused arenas (deterministic layout every call). K padded to 64.
    const size_t OFF_H1   = 0;          //  32 MB  [64,32,64,64]  fp32
    const size_t OFF_BT   = 32 * MB;    // 160 MB  im2col scratch (max, conv2 Kp=320)
    const size_t OFF_H2   = 192 * MB;   //  16 MB  [64,64,32,32]  fp32
    const size_t OFF_CONV = 208 * MB;   // 128 MB  conv out scratch (max, conv1)
    const size_t OFF_H3   = 336 * MB;   //   8 MB  [64,128,16,16] fp32
    size_t bump = 352 * MB;
    auto alloc = [&](size_t bytes) { size_t r = bump; bump += (bytes + 255) & ~(size_t)255; return r; };

    size_t oW1   = alloc((size_t)32 * 64 * 2);
    size_t oW2   = alloc((size_t)64 * 320 * 2);
    size_t oW3   = alloc((size_t)128 * 576 * 2);
    size_t oA1   = alloc(32 * 4),  oB1 = alloc(32 * 4);
    size_t oA2   = alloc(64 * 4),  oB2 = alloc(64 * 4);
    size_t oA3   = alloc(128 * 4), oB3 = alloc(128 * 4);
    size_t oFBF  = alloc((size_t)64 * 32768 * 2);
    size_t oP1W  = alloc((size_t)128 * 32768 * 2);
    size_t oP2W  = alloc((size_t)64 * 128 * 2);
    size_t oP    = alloc((size_t)64 * 128 * 4);
    size_t oPBF  = alloc((size_t)64 * 128 * 2);
    size_t oP2   = alloc((size_t)64 * 64 * 4);
    size_t oXT   = alloc((size_t)1600 * 64 * 2);
    size_t oWih0 = alloc((size_t)2 * 768 * 64 * 2);
    size_t oWhh0 = alloc((size_t)2 * 768 * 256 * 2);
    size_t oWih1 = alloc((size_t)2 * 768 * 512 * 2);
    size_t oWhh1 = alloc((size_t)2 * 768 * 256 * 2);
    size_t oGI   = alloc((size_t)1600 * 768 * 4);
    size_t oGH   = alloc((size_t)64 * 768 * 4);
    size_t oHF   = alloc((size_t)64 * 256 * 4);
    size_t oHB   = alloc((size_t)64 * 256 * 2);
    size_t oY0   = alloc((size_t)1600 * 512 * 4);
    size_t oY0B  = alloc((size_t)1600 * 512 * 2);
    size_t oY1   = alloc((size_t)1600 * 512 * 4);
    size_t oY1B  = alloc((size_t)1600 * 512 * 2);
    size_t oLW   = alloc((size_t)256 * 512 * 2);
    size_t oLIN  = alloc((size_t)1600 * 256 * 4);
    size_t oSPK  = alloc((size_t)1600 * 256 * 4);
    size_t oFT   = alloc((size_t)64 * 768 * 4);
    size_t oFTB  = alloc((size_t)64 * 768 * 2);
    size_t oF1W  = alloc((size_t)512 * 768 * 2);
    size_t oO1   = alloc((size_t)64 * 512 * 4);
    size_t oO1B  = alloc((size_t)64 * 512 * 2);
    size_t oF2W  = alloc((size_t)128 * 512 * 2);
    size_t oO2   = alloc((size_t)64 * 128 * 4);
    size_t oO2B  = alloc((size_t)64 * 128 * 2);
    size_t oF3W  = alloc((size_t)16 * 128 * 2);

    auto FP = [&](size_t off) -> float*  { return (float*)(ws + off); };
    auto BP = [&](size_t off) -> __bf16* { return (__bf16*)(ws + off); };
    auto GRID1 = [](size_t total) { return dim3((unsigned)((total + 255) / 256)); };

    auto pack = [&](const float* src, __bf16* dst, int M, int K, int Mp, int Kp) {
        size_t total = (size_t)Mp * Kp;
        k_pack_bf16<<<GRID1(total), 256, 0, stream>>>(src, dst, M, K, Mp, Kp);
    };
    auto gemm = [&](const __bf16* A, const __bf16* Bt, float* C, __bf16* Cbf,
                    int Mp, int Np, int K, int ldc, int Ms, int Ns,
                    const float* alpha, const float* beta, int act, int mode, int HW) {
        dim3 grid((unsigned)(Np / 64), (unsigned)(Mp / 16));
        dim3 blk(32, 4);
        k_gemm_bf16<<<grid, blk, 0, stream>>>(A, Bt, C, Cbf, K, ldc, Ms, Ns,
                                              alpha, beta, act, mode, HW);
    };

    // ---- conv blocks (explicit im2col + WMMA GEMM + BN/LeakyReLU + pool) ----
    auto conv_block = [&](const float* src, int C, int H, int W, int Cout,
                          size_t wOff, int Kp, size_t aOff, size_t bOff, float* hout) {
        size_t total = (size_t)B * H * W * Kp;
        k_im2col<<<GRID1(total), 256, 0, stream>>>(src, BP(OFF_BT), B, C, H, W, Kp);
        int Np = B * H * W;
        gemm(BP(wOff), BP(OFF_BT), FP(OFF_CONV), nullptr,
             Cout, Np, Kp, 0, Cout, Np, FP(aOff), FP(bOff), /*leaky*/2, /*NCHW*/1, H * W);
        size_t pt = (size_t)B * Cout * (H / 2) * (W / 2);
        k_maxpool2<<<GRID1(pt), 256, 0, stream>>>(FP(OFF_CONV), hout, B * Cout, H, W);
    };

    pack(conv1_w, BP(oW1), 32, 27, 32, 64);      // K 27 -> 64
    pack(conv2_w, BP(oW2), 64, 288, 64, 320);    // K 288 -> 320
    pack(conv3_w, BP(oW3), 128, 576, 128, 576);  // 576 = 9*64
    k_bnparam<<<1, 32, 0, stream>>>(bn1_g, bn1_b, bn1_m, bn1_v, conv1_b, FP(oA1), FP(oB1), 32);
    k_bnparam<<<1, 64, 0, stream>>>(bn2_g, bn2_b, bn2_m, bn2_v, conv2_b, FP(oA2), FP(oB2), 64);
    k_bnparam<<<1, 128, 0, stream>>>(bn3_g, bn3_b, bn3_m, bn3_v, conv3_b, FP(oA3), FP(oB3), 128);

    conv_block(x,          3, 128, 128,  32, oW1,  64, oA1, oB1, FP(OFF_H1));
    conv_block(FP(OFF_H1), 32,  64,  64,  64, oW2, 320, oA2, oB2, FP(OFF_H2));
    conv_block(FP(OFF_H2), 64,  32,  32, 128, oW3, 576, oA3, oB3, FP(OFF_H3));

    // ---- projection head: f[64,32768] -> p[64,128] -> p2[64,64] ----
    pack(FP(OFF_H3), BP(oFBF), 64, 32768, 64, 32768);
    pack(proj1_w, BP(oP1W), 128, 32768, 128, 32768);
    gemm(BP(oP1W), BP(oFBF), FP(oP), BP(oPBF), 128, 64, 32768, 128, 128, 64,
         nullptr, proj1_b, /*relu*/1, 0, 0);
    pack(proj2_w, BP(oP2W), 64, 128, 64, 128);
    gemm(BP(oP2W), BP(oPBF), FP(oP2), nullptr, 64, 64, 128, 64, 64, 64,
         nullptr, proj2_b, 0, 0, 0);
    k_broadcast_time<<<GRID1((size_t)B * T * 64), 256, 0, stream>>>(FP(oP2), BP(oXT), B, T, 64);

    // ---- GRU weights to bf16 (per direction) ----
    pack(gru0_wih,                BP(oWih0),                768,  64, 768,  64);
    pack(gru0_wih + 768 * 64,     BP(oWih0) + 768 * 64,     768,  64, 768,  64);
    pack(gru0_whh,                BP(oWhh0),                768, 256, 768, 256);
    pack(gru0_whh + 768 * 256,    BP(oWhh0) + 768 * 256,    768, 256, 768, 256);
    pack(gru1_wih,                BP(oWih1),                768, 512, 768, 512);
    pack(gru1_wih + 768 * 512,    BP(oWih1) + 768 * 512,    768, 512, 768, 512);
    pack(gru1_whh,                BP(oWhh1),                768, 256, 768, 256);
    pack(gru1_whh + 768 * 256,    BP(oWhh1) + 768 * 256,    768, 256, 768, 256);

    // ---- bidirectional GRU runner ----
    auto run_dir = [&](const __bf16* wihf, const __bf16* whhf,
                       const float* bih, const float* bhh,
                       const __bf16* xbf, int In, float* ybuf, int dir) {
        // gi = x @ wih^T + bih  : [1600, 768]
        gemm(wihf, xbf, FP(oGI), nullptr, 768, B * T, In, 768, 768, B * T,
             nullptr, bih, 0, 0, 0);
        k_fill_f32 <<<GRID1((size_t)B * 256), 256, 0, stream>>>(FP(oHF), (size_t)B * 256, 0.f);
        k_fill_bf16<<<GRID1((size_t)B * 256), 256, 0, stream>>>(BP(oHB), (size_t)B * 256, 0.f);
        for (int t = 0; t < T; ++t) {
            int tt = dir == 0 ? t : (T - 1 - t);
            // gh = h @ whh^T + bhh : [64, 768]
            gemm(whhf, BP(oHB), FP(oGH), nullptr, 768, B, 256, 768, 768, B,
                 nullptr, bhh, 0, 0, 0);
            k_gru_step<<<GRID1((size_t)B * 256), 256, 0, stream>>>(
                FP(oGI), FP(oGH), FP(oHF), BP(oHB), ybuf, tt, B, 256, dir, T);
        }
    };

    run_dir(BP(oWih0),            BP(oWhh0),             gru0_bih,       gru0_bhh,
            BP(oXT), 64, FP(oY0), 0);
    run_dir(BP(oWih0) + 768 * 64, BP(oWhh0) + 768 * 256, gru0_bih + 768, gru0_bhh + 768,
            BP(oXT), 64, FP(oY0), 1);

    pack(FP(oY0), BP(oY0B), B * T, 512, B * T, 512);
    run_dir(BP(oWih1),             BP(oWhh1),             gru1_bih,       gru1_bhh,
            BP(oY0B), 512, FP(oY1), 0);
    run_dir(BP(oWih1) + 768 * 512, BP(oWhh1) + 768 * 256, gru1_bih + 768, gru1_bhh + 768,
            BP(oY0B), 512, FP(oY1), 1);

    // ---- LIF branch ----
    pack(FP(oY1), BP(oY1B), B * T, 512, B * T, 512);
    pack(lifproj_w, BP(oLW), 256, 512, 256, 512);
    gemm(BP(oLW), BP(oY1B), FP(oLIN), nullptr, 256, B * T, 512, 256, 256, B * T,
         nullptr, lifproj_b, 0, 0, 0);
    k_lif<<<GRID1((size_t)B * 256), 256, 0, stream>>>(FP(oLIN), FP(oSPK), B, T, 256);

    // ---- fusion + FC head ----
    k_fusion<<<GRID1((size_t)B * 768), 256, 0, stream>>>(FP(oY1), FP(oSPK),
                                                         FP(oFT), BP(oFTB), B, T);
    pack(fc1_w, BP(oF1W), 512, 768, 512, 768);
    gemm(BP(oF1W), BP(oFTB), FP(oO1), BP(oO1B), 512, 64, 768, 512, 512, 64,
         nullptr, fc1_b, 1, 0, 0);
    pack(fc2_w, BP(oF2W), 128, 512, 128, 512);
    gemm(BP(oF2W), BP(oO1B), FP(oO2), BP(oO2B), 128, 64, 512, 128, 128, 64,
         nullptr, fc2_b, 1, 0, 0);
    pack(fc3_w, BP(oF3W), 2, 128, 16, 128);   // pad M 2 -> 16 for WMMA
    gemm(BP(oF3W), BP(oO2B), (float*)d_out, nullptr, 16, 64, 128, 2, 2, 64,
         nullptr, fc3_b, 0, 0, 0);
}
